// feature_extraction_32968168964590
// MI455X (gfx1250) — matrix operand order repaired and checked by reading, not compiled, not run on hardware
//
#include <hip/hip_runtime.h>

typedef __attribute__((ext_vector_type(16))) _Float16 v16h;
typedef __attribute__((ext_vector_type(8)))  _Float16 v8h;
typedef __attribute__((ext_vector_type(8)))  float    v8f;

#define TPB 256

static inline int cdiv_h(long a, long b) { return (int)((a + b - 1) / b); }

// ---------------- device kernels ----------------

__global__ void k_zero_f32(float* __restrict__ p, int n) {
  int i = blockIdx.x * blockDim.x + threadIdx.x;
  if (i < n) p[i] = 0.f;
}

// stage x (n x cin, f32) -> padded (Mp x Kp, f16), zero fill
__global__ void k_stage_a(const float* __restrict__ x, _Float16* __restrict__ xh,
                          int n, int cin, int Kp, int total) {
  int i = blockIdx.x * blockDim.x + threadIdx.x;
  if (i >= total) return;
  int k = i % Kp;
  int m = i / Kp;
  float v = (m < n && k < cin) ? x[m * cin + k] : 0.f;
  xh[i] = (_Float16)v;
}

// stage weight (cin x N, f32) -> TRANSPOSED padded (N x Kp, f16) so that each
// lane's WMMA B-fragment is 16 contiguous halves (two b128 loads).
__global__ void k_stage_bt(const float* __restrict__ w, _Float16* __restrict__ wt,
                           int cin, int N, int Kp, int total) {
  int i = blockIdx.x * blockDim.x + threadIdx.x;
  if (i >= total) return;            // total = N * Kp
  int k = i % Kp;
  int c = i / Kp;
  float v = (k < cin) ? w[k * N + c] : 0.f;
  wt[i] = (_Float16)v;
}

// C[n x N] = A[Mp x Kp] * Bt[N x Kp]^T. One wave per (16 x 16*NT) tile strip,
// 4 waves per block. f16 WMMA, f32 accumulate.
template <int NT>
__global__ void k_gemm_wmma(const _Float16* __restrict__ A, const _Float16* __restrict__ Bt,
                            float* __restrict__ C, int n, int N, int Kp, int tilesM) {
  int lane  = threadIdx.x & 31;
  int wave  = threadIdx.x >> 5;
  int mtile = blockIdx.y * 4 + wave;
  if (mtile >= tilesM) return;       // wave-uniform exit
  int lo = lane & 15;
  int hi = lane >> 4;
  int nt = blockIdx.x * (16 * NT);
  int mt = mtile * 16;

  const _Float16* arow = A + (size_t)(mt + lo) * Kp;
  const _Float16* brow[NT];
#pragma unroll
  for (int t = 0; t < NT; ++t) brow[t] = Bt + (size_t)(nt + 16 * t + lo) * Kp;

  __builtin_prefetch(arow, 0, 3);            // global_prefetch_b8
  __builtin_prefetch(brow[0], 0, 3);

  v8f acc[NT];
#pragma unroll
  for (int t = 0; t < NT; ++t) acc[t] = (v8f){};

  for (int kb = 0; kb < Kp; kb += 32) {
    // A 16x32 f16 fragment: lane row M=lo; halves 0..15 -> K=kb+16*hi+i
    // (contiguous, matching B). Two contiguous 16B chunks.
    v16h a;
    *(v8h*)&a       = *(const v8h*)(arow + kb + 16 * hi);
    *((v8h*)&a + 1) = *(const v8h*)(arow + kb + 16 * hi + 8);
#pragma unroll
    for (int t = 0; t < NT; ++t) {
      // B 32x16 fragment: lane column N=nt+16t+lo; K = kb+16*hi+0..15, contiguous
      // in the transposed staging.
      v16h b;
      *(v8h*)&b       = *(const v8h*)(brow[t] + kb + 16 * hi);
      *((v8h*)&b + 1) = *(const v8h*)(brow[t] + kb + 16 * hi + 8);
      acc[t] = __builtin_amdgcn_wmma_f32_16x16x32_f16(false, a, false, b,
                                                      (short)0, acc[t], false, false);
    }
  }

  // C/D layout: VGPR r holds (M = mt + r + 8*hi, N = col). All-int index math.
  int m0 = mt + 8 * hi;
#pragma unroll
  for (int t = 0; t < NT; ++t) {
    int col = nt + 16 * t + lo;
    if (m0 + 7 < n) {
#pragma unroll
      for (int r = 0; r < 8; ++r) C[(m0 + r) * N + col] = acc[t][r];
    } else {
#pragma unroll
      for (int r = 0; r < 8; ++r)
        if (m0 + r < n) C[(m0 + r) * N + col] = acc[t][r];
    }
  }
}

// Gaussian kernel weights w[e,k] = exp(-0.5 * sum_d (pseudo-mu)^2/(eps+sigma^2))
__global__ void k_edge_w(const float* __restrict__ pseudo, const float* __restrict__ mu,
                         const float* __restrict__ sigma, float* __restrict__ w, int E) {
  int i = blockIdx.x * blockDim.x + threadIdx.x;
  if (i >= E * 10) return;
  int e = i / 10, k = i - e * 10;
  float dx = pseudo[2 * e]     - mu[2 * k];
  float dy = pseudo[2 * e + 1] - mu[2 * k + 1];
  float sx = sigma[2 * k], sy = sigma[2 * k + 1];
  float t = -0.5f * (dx * dx / (1e-15f + sx * sx) + dy * dy / (1e-15f + sy * sy));
  w[i] = expf(t);
}

// per (edge, feature): msg = sum_k h[src][k][f]*w[e][k]; scatter-add to agg[dst][f].
__global__ void k_scatter(const float* __restrict__ h, const float* __restrict__ w,
                          const int* __restrict__ src, const int* __restrict__ dst,
                          float* __restrict__ agg, float* __restrict__ cnt,
                          int E, int cout) {
  int i = blockIdx.x * blockDim.x + threadIdx.x;
  if (i >= E * cout) return;
  int e = i / cout;
  int f = i - e * cout;
  int s = src[e], d = dst[e];
  const float* hp = h + s * (10 * cout) + f;
  const float* wp = w + e * 10;
  float t = 0.f;
#pragma unroll
  for (int k = 0; k < 10; ++k) t += hp[k * cout] * wp[k];
  atomicAdd(agg + d * cout + f, t);
  if (f == 0) atomicAdd(cnt + d, 1.f);
}

// out = leaky( agg/max(cnt,1) + hroot + bias )
__global__ void k_finalize(const float* __restrict__ agg, const float* __restrict__ cnt,
                           const float* __restrict__ hroot, const float* __restrict__ bias,
                           float* __restrict__ out, int n, int cout) {
  int i = blockIdx.x * blockDim.x + threadIdx.x;
  if (i >= n * cout) return;
  int f    = i % cout;
  int node = i / cout;
  float c = cnt[node];
  c = c > 1.f ? c : 1.f;
  float v = agg[i] / c + hroot[i] + bias[f];
  out[i] = v > 0.f ? v : 0.2f * v;
}

// Reproduces the reference's raw-reinterpret reshape: (n2,7,C) viewed as (n2,C,7),
// max over last axis. Flat block index t = c*7+j maps to neighbor t/C, channel t%C.
__global__ void k_hexpool(const float* __restrict__ x, const int* __restrict__ hexa,
                          float* __restrict__ xp, int n2, int C) {
  int i = blockIdx.x * blockDim.x + threadIdx.x;
  if (i >= n2 * C) return;
  int c    = i % C;
  int node = i / C;
  const int* hrow = hexa + node * 7;
  float m = -3.4e38f;
#pragma unroll
  for (int j = 0; j < 7; ++j) {
    int t = c * 7 + j;
    int row = hrow[t / C];
    float v = x[row * C + (t % C)];
    m = fmaxf(m, v);
  }
  xp[i] = m;
}

// out[i] = [ x[i,0:C] | xp[i,0:C] | inp0[i,0:2] ]
__global__ void k_concat(const float* __restrict__ x, const float* __restrict__ xp,
                         const float* __restrict__ inp0, float* __restrict__ out,
                         int n2, int C) {
  int Ct = 2 * C + 2;
  int i = blockIdx.x * blockDim.x + threadIdx.x;
  if (i >= n2 * Ct) return;
  int c    = i % Ct;
  int node = i / Ct;
  float v;
  if (c < C)          v = x[node * C + c];
  else if (c < 2 * C) v = xp[node * C + (c - C)];
  else                v = inp0[node * 2 + (c - 2 * C)];
  out[i] = v;
}

// ---------------- host orchestration ----------------

extern "C" void kernel_launch(void* const* d_in, const int* in_sizes, int n_in,
                              void* d_out, int out_size, void* d_ws, size_t ws_size,
                              hipStream_t stream) {
  (void)in_sizes; (void)n_in; (void)out_size; (void)ws_size;

  static const int DIMS_[5]  = {40962, 10242, 2562, 642, 162};
  static const int EDGES_[5] = {245760, 61440, 15360, 3840, 960};
  static const int FEATS_[5] = {32, 64, 128, 256, 512};

  const float* moving = (const float*)d_in[0];
  const float* target = (const float*)d_in[1];
  const int*   edge[5];
  const float* pseudo[5];
  const int*   hexa[4];
  for (int l = 0; l < 5; ++l) edge[l]   = (const int*)d_in[2 + l];
  for (int l = 0; l < 5; ++l) pseudo[l] = (const float*)d_in[7 + l];
  for (int l = 0; l < 4; ++l) hexa[l]   = (const int*)d_in[12 + l];

  // params pytree: 16 sets (sorted: s4a s4b s5a s5b x1a x1b x2a x2b x3a x3b
  // y1a y1b y2a y2b y3a y3b), each with leaves sorted: bias, g, mu, root, sigma.
  struct P { const float *bias, *g, *mu, *root, *sigma; };
  P ps[16];
  for (int s = 0; s < 16; ++s) {
    ps[s].bias  = (const float*)d_in[16 + 5 * s + 0];
    ps[s].g     = (const float*)d_in[16 + 5 * s + 1];
    ps[s].mu    = (const float*)d_in[16 + 5 * s + 2];
    ps[s].root  = (const float*)d_in[16 + 5 * s + 3];
    ps[s].sigma = (const float*)d_in[16 + 5 * s + 4];
  }
  const int xsets[10] = {4, 5, 6, 7, 8, 9, 0, 1, 2, 3};      // x1a..x3b, s4a..s5b
  const int ysets[10] = {10, 11, 12, 13, 14, 15, 0, 1, 2, 3};

  // ---- carve workspace ----
  char*  base = (char*)d_ws;
  size_t off  = 0;
  auto carve = [&](size_t bytes) -> char* {
    char* p = base + off;
    off += (bytes + 255) & ~(size_t)255;
    return p;
  };
  float*    bufA  = (float*)carve((size_t)40962 * 66 * 4);   // x ping
  float*    bufB  = (float*)carve((size_t)40962 * 66 * 4);   // x pong
  float*    bufP  = (float*)carve((size_t)10242 * 64 * 4);   // pool out
  float*    hbuf  = (float*)carve((size_t)40962 * 320 * 4);  // h = x@g
  float*    hroot = (float*)carve((size_t)40962 * 32 * 4);   // x@root
  float*    agg   = (float*)carve((size_t)40962 * 32 * 4);
  float*    cnt   = (float*)carve((size_t)40962 * 4);
  float*    wbuf  = (float*)carve((size_t)245760 * 10 * 4);  // edge weights
  _Float16* xh    = (_Float16*)carve((size_t)1500000 * 2);   // staged A (Mp x Kp)
  _Float16* gh    = (_Float16*)carve((size_t)3000000 * 2);   // staged g^T (Ng x Kp)
  _Float16* rh    = (_Float16*)carve((size_t)320000 * 2);    // staged root^T

  auto run_conv = [&](const float* x, int n, int cin, int cout, const P& p,
                      const int* ed, const float* psd, int E, float* out) {
    int Mp = ((n + 15) / 16) * 16;
    int Kp = ((cin + 31) / 32) * 32;
    int Ng = 10 * cout;
    int tilesM = Mp / 16;
    int gy = (tilesM + 3) / 4;

    int ta = Mp * Kp;
    k_stage_a<<<cdiv_h(ta, TPB), TPB, 0, stream>>>(x, xh, n, cin, Kp, ta);
    int tg = Ng * Kp;
    k_stage_bt<<<cdiv_h(tg, TPB), TPB, 0, stream>>>(p.g, gh, cin, Ng, Kp, tg);
    int tr = cout * Kp;
    k_stage_bt<<<cdiv_h(tr, TPB), TPB, 0, stream>>>(p.root, rh, cin, cout, Kp, tr);

    if (Ng % 64 == 0)
      k_gemm_wmma<4><<<dim3(Ng / 64, gy), 128, 0, stream>>>(xh, gh, hbuf, n, Ng, Kp, tilesM);
    else
      k_gemm_wmma<2><<<dim3(Ng / 32, gy), 128, 0, stream>>>(xh, gh, hbuf, n, Ng, Kp, tilesM);
    if (cout % 64 == 0)
      k_gemm_wmma<4><<<dim3(cout / 64, gy), 128, 0, stream>>>(xh, rh, hroot, n, cout, Kp, tilesM);
    else
      k_gemm_wmma<2><<<dim3(cout / 32, gy), 128, 0, stream>>>(xh, rh, hroot, n, cout, Kp, tilesM);

    k_edge_w<<<cdiv_h((long)E * 10, TPB), TPB, 0, stream>>>(psd, p.mu, p.sigma, wbuf, E);
    k_zero_f32<<<cdiv_h((long)n * cout, TPB), TPB, 0, stream>>>(agg, n * cout);
    k_zero_f32<<<cdiv_h(n, TPB), TPB, 0, stream>>>(cnt, n);
    k_scatter<<<cdiv_h((long)E * cout, TPB), TPB, 0, stream>>>(hbuf, wbuf, ed, ed + E,
                                                               agg, cnt, E, cout);
    k_finalize<<<cdiv_h((long)n * cout, TPB), TPB, 0, stream>>>(agg, cnt, hroot,
                                                                p.bias, out, n, cout);
  };

  auto run_branch = [&](const float* inp0, const int* sets, float* outdst) {
    const float* xin = inp0;
    int cin = 2;
    float *A = bufA, *B = bufB;
    for (int lvl = 0; lvl < 5; ++lvl) {
      int n = DIMS_[lvl], E = EDGES_[lvl], F = FEATS_[lvl];
      run_conv(xin, n, cin, F, ps[sets[2 * lvl]],     edge[lvl], pseudo[lvl], E, A);
      run_conv(A,   n, F,   F, ps[sets[2 * lvl + 1]], edge[lvl], pseudo[lvl], E, B);
      if (lvl < 4) {
        int n2 = DIMS_[lvl + 1];
        k_hexpool<<<cdiv_h((long)n2 * F, TPB), TPB, 0, stream>>>(B, hexa[lvl], bufP, n2, F);
        k_concat<<<cdiv_h((long)n2 * (2 * F + 2), TPB), TPB, 0, stream>>>(B, bufP, inp0,
                                                                          A, n2, F);
        xin = A;
        cin = 2 * F + 2;
        float* t = A; A = B; B = t;
      } else {
        hipMemcpyAsync(outdst, B, (size_t)162 * 512 * 4,
                       hipMemcpyDeviceToDevice, stream);
      }
    }
  };

  float* out = (float*)d_out;
  run_branch(moving, xsets, out);
  run_branch(target, ysets, out + (size_t)162 * 512);
}
